// MultiHeadFAIIA_3874060501114
// MI455X (gfx1250) — matrix operand order repaired and checked by reading, not compiled, not run on hardware
//
#include <hip/hip_runtime.h>
#include <hip/hip_bf16.h>
#include <math.h>

// MultiHeadFAIIA fused kernel for gfx1250 (MI455X).
// B=262144, D=256, H=4, A=32, P=4, r=64. All inputs f32, output f32.
// v2: one-time weight transpose/convert prep kernel into d_ws (f16), then the
// fused kernel pulls weight tiles with GLOBAL_LOAD_ASYNC_TO_LDS_B128
// (ASYNCcnt) overlapped with compute. GEMMs via v_wmma_f32_16x16x32_f16.

#define SCALE_ATT 0.1767766953f   // 32^-0.5
#define EPS_LN    1e-5f

typedef __attribute__((ext_vector_type(16))) _Float16 v16h;
typedef __attribute__((ext_vector_type(8)))  _Float16 v8h;
typedef __attribute__((ext_vector_type(8)))  float    v8f;

union Frag16 { v16h v; v8h h2[2]; };

#define XS  264   // stride for K=256 f16 tiles (+8 pad halfs)
#define WS2 136   // stride for K=128 f16 tiles
#define WS4 72    // stride for K=64  f16 tiles

// d_ws layout (f16 elements): pre-transposed weights
#define WSOFF_WQ  0        // WqT [128][256]
#define WSOFF_WF  32768    // WfT [256][128]
#define WSOFF_G1  65536    // Wg1T [64][256] (rows 0..255 of Wg1)
#define WSOFF_G2  81920    // Wg2T [256][64]
#define WS_HALFS  98304    // total f16 elements

struct __align__(16) Smem {
  _Float16 xtile[128 * XS];    // x tile, f16 row-major           67584 B
  _Float16 wbuf[256 * WS2];    // WqT then WfT                    69632 B
  _Float16 qbuf[128 * WS2];    // q = x@Wq + bq; later Wg2T       34816 B
  _Float16 cbuf[128 * WS2];    // combined; later Wg1T            34816 B
  _Float16 hbuf[128 * WS4];    // gate hidden (post relu)         18432 B
  _Float16 outb[128 * XS];     // out (pre-gate), then y          67584 B
  float fw[128 * 4];           // focal weight; later LN partials
  float unc[128];              // uncertainty == difficulty
  float bqS[128], boS[128], lngS[128], lnbS[128];
  float pkS[512], pvS[512], pimpS[16];
  float WoS[4096];
  float bfS[256], bg2S[256], lnfgS[256], lnfbS[256];
  float g1last[64], bg1S[64];
  float muS[128], rsS[128];
};

// A-fragment (16x32 f16) per ISA layout: lanes 0-15 rows M, e[0..7]=K kk..kk+7,
// e[8..15]=K kk+16..kk+23, with kk = k0 + (lane>=16 ? 8 : 0).
__device__ __forceinline__ v16h frag_a(const _Float16* __restrict__ base,
                                       int row, int stride, int k0, int lane) {
  Frag16 f;
  const _Float16* p = base + row * stride + k0 + ((lane & 16) ? 8 : 0);
  f.h2[0] = *(const v8h*)(p);
  f.h2[1] = *(const v8h*)(p + 16);
  return f.v;
}

// B-fragment (32x16 f16) from transposed weights WT[n][k]: lane holds column
// n0+(lane&15); e[0..7]=K kk..kk+7, e[8..15]=K kk+16..kk+23, kk = k0 + (lane>=16 ? 8 : 0).
__device__ __forceinline__ v16h frag_b(const _Float16* __restrict__ base,
                                       int n0, int stride, int k0, int lane) {
  Frag16 f;
  const _Float16* p = base + (n0 + (lane & 15)) * stride + k0 + ((lane & 16) ? 8 : 0);
  f.h2[0] = *(const v8h*)(p);
  f.h2[1] = *(const v8h*)(p + 16);
  return f.v;
}

#define WMMA_F16(a, b, c) \
  __builtin_amdgcn_wmma_f32_16x16x32_f16(false, (a), false, (b), (short)0, (c), false, false)

// Async global->LDS copy (CDNA5, tracked by ASYNCcnt). lds_off is the 32-bit
// LDS byte address (flat addr truncated), gaddr a 64-bit global address.
__device__ __forceinline__ void async_ld_b128(unsigned int lds_off,
                                              const _Float16* gaddr) {
  asm volatile("global_load_async_to_lds_b128 %0, %1, off"
               :: "v"(lds_off), "v"(gaddr) : "memory");
}
__device__ __forceinline__ void wait_async0() {
  asm volatile("s_wait_asynccnt 0x0" ::: "memory");
}
__device__ __forceinline__ unsigned int lds_off32(const void* p) {
  return (unsigned int)(unsigned long long)p;   // addr[31:0] == LDS offset
}

// ---------------- one-time weight transpose/convert into d_ws --------------
__global__ __launch_bounds__(256) void prep_weights(
    const float* __restrict__ Wq, const float* __restrict__ Wf,
    const float* __restrict__ Wg1, const float* __restrict__ Wg2,
    _Float16* __restrict__ ws) {
  int idx = blockIdx.x * 256 + threadIdx.x;
  if (idx < 32768) {                        // WqT[n][k] = Wq[h][k][a]
    int n = idx >> 8, k = idx & 255, h = n >> 5, a = n & 31;
    ws[WSOFF_WQ + idx] = (_Float16)Wq[(h << 13) + (k << 5) + a];
  } else if (idx < 65536) {                 // WfT[n][k] = Wf[k][n]
    int j = idx - 32768; int n = j >> 7, k = j & 127;
    ws[idx] = (_Float16)Wf[(k << 8) + n];
  } else if (idx < 81920) {                 // Wg1T[n][k] = Wg1[k][n]
    int j = idx - 65536; int n = j >> 8, k = j & 255;
    ws[idx] = (_Float16)Wg1[(k << 6) + n];
  } else if (idx < WS_HALFS) {              // Wg2T[n][k] = Wg2[k][n]
    int j = idx - 81920; int n = j >> 6, k = j & 63;
    ws[idx] = (_Float16)Wg2[(k << 8) + n];
  }
}

template <int ASYNC>
__global__ __launch_bounds__(256) void faiia_fused(
    const float* __restrict__ x,    const float* __restrict__ mp,
    const float* __restrict__ Wq,   const float* __restrict__ bq,
    const float* __restrict__ pk,   const float* __restrict__ pv,
    const float* __restrict__ pimp, const float* __restrict__ falpha,
    const float* __restrict__ ftemp,const float* __restrict__ Wo,
    const float* __restrict__ bo,   const float* __restrict__ lng,
    const float* __restrict__ lnb,  const float* __restrict__ Wf,
    const float* __restrict__ bf,   const float* __restrict__ Wg1,
    const float* __restrict__ bg1,  const float* __restrict__ Wg2,
    const float* __restrict__ bg2,  const float* __restrict__ lnfg,
    const float* __restrict__ lnfb, const _Float16* __restrict__ wsw,
    float* __restrict__ out)
{
  __shared__ Smem sm;
  const int tid  = threadIdx.x;
  const int lane = tid & 31;
  const int wave = tid >> 5;
  const int wr0  = wave * 16;
  const int rsel = (lane & 16) ? 8 : 0;
  const int csel = lane & 15;
  const int rowbase = blockIdx.x * 128;

  // ---------------- P0: kick off WqT async copy, stage x tile, focal terms,
  //                      small params ----------------
  {
    if constexpr (ASYNC) {
      // WqT: 128 rows x 32 b128-chunks, packed global -> padded LDS
      for (int idx = tid; idx < 4096; idx += 256) {
        int n = idx >> 5, kc = (idx & 31) << 3;
        async_ld_b128(lds_off32(&sm.wbuf[n * XS + kc]),
                      wsw + WSOFF_WQ + (n << 8) + kc);
      }
    }
    const float4* x4 = (const float4*)(x + (size_t)rowbase * 256);
    for (int i = tid; i < 128 * 64; i += 256) {            // 128 rows x 64 float4
      int r = i >> 6, c4 = (i & 63) << 2;
      float4 v = x4[i];
      _Float16* p = &sm.xtile[r * XS + c4];
      p[0] = (_Float16)v.x; p[1] = (_Float16)v.y;
      p[2] = (_Float16)v.z; p[3] = (_Float16)v.w;
    }
    if (blockIdx.x + 1 < gridDim.x) {   // speculative prefetch of next tile
      const char* nx = (const char*)(x + (size_t)(rowbase + 128) * 256);
      __builtin_prefetch(nx + tid * 512, 0, 1);
    }
    for (int r = tid; r < 128; r += 256) {
      float m = mp[rowbase + r];
      float u = 1.0f - fabsf(m - 0.5f) * 2.0f;
      sm.unc[r] = u;
      float ug = (u + 1e-8f); ug = ug * ug;                // GAMMA = 2
      for (int h = 0; h < 4; ++h) sm.fw[r * 4 + h] = falpha[h] * ug * ftemp[h];
    }
    for (int i = tid; i < 128; i += 256) {
      sm.bqS[i] = bq[i]; sm.boS[i] = bo[i];
      sm.lngS[i] = lng[i]; sm.lnbS[i] = lnb[i];
    }
    for (int i = tid; i < 512; i += 256) { sm.pkS[i] = pk[i]; sm.pvS[i] = pv[i]; }
    if (tid < 16) sm.pimpS[tid] = pimp[tid];
    for (int i = tid; i < 4096; i += 256) sm.WoS[i] = Wo[i];
    if (tid < 256) {
      sm.bfS[tid] = bf[tid]; sm.bg2S[tid] = bg2[tid];
      sm.lnfgS[tid] = lnfg[tid]; sm.lnfbS[tid] = lnfb[tid];
    }
    if (tid < 64) { sm.g1last[tid] = Wg1[256 * 64 + tid]; sm.bg1S[tid] = bg1[tid]; }
    if constexpr (!ASYNC) {
      for (int idx = tid; idx < 128 * 256; idx += 256) {
        int n = idx >> 8, k = idx & 255;
        int h = n >> 5, a = n & 31;
        sm.wbuf[n * XS + k] = (_Float16)Wq[(h << 13) + (k << 5) + a];
      }
    }
  }
  if constexpr (ASYNC) wait_async0();
  __syncthreads();

  // ---------------- P2: GEMM1  q[128x128] = xtile @ WqT^T + bq -------------
  {
    v16h afr[8];
    #pragma unroll
    for (int kb = 0; kb < 8; ++kb)
      afr[kb] = frag_a(sm.xtile, wr0 + (lane & 15), XS, kb * 32, lane);
    for (int nb = 0; nb < 8; ++nb) {
      v8f acc = {0.f,0.f,0.f,0.f,0.f,0.f,0.f,0.f};
      #pragma unroll
      for (int kb = 0; kb < 8; ++kb) {
        v16h bfr = frag_b(sm.wbuf, nb * 16, XS, kb * 32, lane);
        acc = WMMA_F16(afr[kb], bfr, acc);
      }
      int col = nb * 16 + csel;
      float bias = sm.bqS[col];
      #pragma unroll
      for (int j = 0; j < 8; ++j)
        sm.qbuf[(wr0 + rsel + j) * WS2 + col] = (_Float16)(acc[j] + bias);
    }
  }
  __syncthreads();

  // ---------------- P3: kick off WfT async copy (wbuf free now), then
  //                      proto attention + Wo + per-head LN -> cbuf ---------
  {
    if constexpr (ASYNC) {
      // WfT: 256 rows x 16 b128-chunks
      for (int idx = tid; idx < 4096; idx += 256) {
        int n = idx >> 4, kc = (idx & 15) << 3;
        async_ld_b128(lds_off32(&sm.wbuf[n * WS2 + kc]),
                      wsw + WSOFF_WF + (n << 7) + kc);
      }
    }
    for (int it = 0; it < 2; ++it) {
      int p = it * 256 + tid;
      int row = p & 127, head = p >> 7;
      float qv[32];
      #pragma unroll
      for (int a = 0; a < 32; ++a)
        qv[a] = (float)sm.qbuf[row * WS2 + head * 32 + a];
      float fwm = 1.0f + sm.fw[row * 4 + head];
      float s[4], smax = -1e30f;
      #pragma unroll
      for (int pi = 0; pi < 4; ++pi) {
        float d = sm.pimpS[head * 4 + pi];
        const float* pkp = &sm.pkS[(head * 4 + pi) * 32];
        #pragma unroll
        for (int a = 0; a < 32; ++a) d += qv[a] * pkp[a];
        d = d * fwm * SCALE_ATT;
        s[pi] = d; smax = fmaxf(smax, d);
      }
      float den = 0.f;
      #pragma unroll
      for (int pi = 0; pi < 4; ++pi) { s[pi] = __expf(s[pi] - smax); den += s[pi]; }
      float inv = 1.0f / den;
      float att[32];
      #pragma unroll
      for (int a = 0; a < 32; ++a) att[a] = 0.f;
      #pragma unroll
      for (int pi = 0; pi < 4; ++pi) {
        float w = s[pi] * inv;
        const float* pvp = &sm.pvS[(head * 4 + pi) * 32];
        #pragma unroll
        for (int a = 0; a < 32; ++a) att[a] += w * pvp[a];
      }
      float ho[32];
      #pragma unroll
      for (int o = 0; o < 32; ++o) ho[o] = sm.boS[head * 32 + o];
      #pragma unroll
      for (int a = 0; a < 32; ++a) {
        float av = att[a];
        const float* wrow = &sm.WoS[(head * 32 + a) * 32];
        #pragma unroll
        for (int o = 0; o < 32; ++o) ho[o] += av * wrow[o];
      }
      float m = 0.f, v2 = 0.f;
      #pragma unroll
      for (int o = 0; o < 32; ++o) { m += ho[o]; v2 += ho[o] * ho[o]; }
      m *= (1.0f / 32.0f); v2 = v2 * (1.0f / 32.0f) - m * m;
      float rst = rsqrtf(v2 + EPS_LN);
      #pragma unroll
      for (int o = 0; o < 32; ++o) {
        float y = (ho[o] - m) * rst * sm.lngS[head * 32 + o] + sm.lnbS[head * 32 + o];
        sm.cbuf[row * WS2 + head * 32 + o] = (_Float16)y;
      }
    }
    if constexpr (!ASYNC) {
      for (int idx = tid; idx < 256 * 128; idx += 256) {
        int n = idx >> 7, k = idx & 127;
        sm.wbuf[n * WS2 + k] = (_Float16)Wf[(k << 8) + n];
      }
    }
  }
  if constexpr (ASYNC) wait_async0();
  __syncthreads();

  // ---------------- P4: preload combined A-frags; stage Wg1T into cbuf
  //                      region + Wg2T into qbuf region (both dead);
  //                      GEMM2 out = cbuf @ WfT^T + bf -> outb --------------
  _Float16* g1t = sm.cbuf;   // Wg1T, stride XS (64*264 = 16896 <= 17408)
  _Float16* g2t = sm.qbuf;   // Wg2T, stride WS4 (256*72 = 18432 <= 17408*2)
  {
    v16h afr[4];
    #pragma unroll
    for (int kb = 0; kb < 4; ++kb)
      afr[kb] = frag_a(sm.cbuf, wr0 + (lane & 15), WS2, kb * 32, lane);
    __syncthreads();   // all A-frags in regs before cbuf/qbuf are reused
    if constexpr (ASYNC) {
      for (int idx = tid; idx < 2048; idx += 256) {        // Wg1T 64x32 chunks
        int n = idx >> 5, kc = (idx & 31) << 3;
        async_ld_b128(lds_off32(&g1t[n * XS + kc]),
                      wsw + WSOFF_G1 + (n << 8) + kc);
      }
      for (int idx = tid; idx < 2048; idx += 256) {        // Wg2T 256x8 chunks
        int n = idx >> 3, kc = (idx & 7) << 3;
        async_ld_b128(lds_off32(&g2t[n * WS4 + kc]),
                      wsw + WSOFF_G2 + (n << 6) + kc);
      }
    } else {
      for (int idx = tid; idx < 64 * 256; idx += 256) {
        int n = idx >> 8, k = idx & 255;
        g1t[n * XS + k] = (_Float16)Wg1[(k << 6) + n];
      }
      for (int idx = tid; idx < 256 * 64; idx += 256) {
        int n = idx >> 6, k = idx & 63;
        g2t[n * WS4 + k] = (_Float16)Wg2[(k << 8) + n];
      }
    }
    for (int nb = 0; nb < 16; ++nb) {
      v8f acc = {0.f,0.f,0.f,0.f,0.f,0.f,0.f,0.f};
      #pragma unroll
      for (int kb = 0; kb < 4; ++kb) {
        v16h bfr = frag_b(sm.wbuf, nb * 16, WS2, kb * 32, lane);
        acc = WMMA_F16(afr[kb], bfr, acc);
      }
      int col = nb * 16 + csel;
      float bias = sm.bfS[col];
      #pragma unroll
      for (int j = 0; j < 8; ++j)
        sm.outb[(wr0 + rsel + j) * XS + col] = (_Float16)(acc[j] + bias);
    }
  }
  if constexpr (ASYNC) wait_async0();
  __syncthreads();

  // ---------------- P6: gate GEMM1  hidden = relu(out@Wg1 + diff*row256 + bg1)
  {
    v16h afr[8];
    #pragma unroll
    for (int kb = 0; kb < 8; ++kb)
      afr[kb] = frag_a(sm.outb, wr0 + (lane & 15), XS, kb * 32, lane);
    for (int nb = 0; nb < 4; ++nb) {
      v8f acc = {0.f,0.f,0.f,0.f,0.f,0.f,0.f,0.f};
      #pragma unroll
      for (int kb = 0; kb < 8; ++kb) {
        v16h bfr = frag_b(g1t, nb * 16, XS, kb * 32, lane);
        acc = WMMA_F16(afr[kb], bfr, acc);
      }
      int col = nb * 16 + csel;
      float g1l = sm.g1last[col], b1 = sm.bg1S[col];
      #pragma unroll
      for (int j = 0; j < 8; ++j) {
        int row = wr0 + rsel + j;
        float v = acc[j] + sm.unc[row] * g1l + b1;
        sm.hbuf[row * WS4 + col] = (_Float16)(v > 0.f ? v : 0.f);
      }
    }
  }
  __syncthreads();

  // ---------------- P8: gate GEMM2 + sigmoid + gate + residual -> outb -----
  {
    v16h afr[2];
    #pragma unroll
    for (int kb = 0; kb < 2; ++kb)
      afr[kb] = frag_a(sm.hbuf, wr0 + (lane & 15), WS4, kb * 32, lane);
    for (int nb = 0; nb < 16; ++nb) {
      v8f acc = {0.f,0.f,0.f,0.f,0.f,0.f,0.f,0.f};
      #pragma unroll
      for (int kb = 0; kb < 2; ++kb) {
        v16h bfr = frag_b(g2t, nb * 16, WS4, kb * 32, lane);
        acc = WMMA_F16(afr[kb], bfr, acc);
      }
      int col = nb * 16 + csel;
      float b2 = sm.bg2S[col];
      #pragma unroll
      for (int j = 0; j < 8; ++j) {
        int row = wr0 + rsel + j;
        float g = 1.0f / (1.0f + __expf(-(acc[j] + b2)));
        float o = (float)sm.outb[row * XS + col];
        float y = o * g + (float)sm.xtile[row * XS + col];
        sm.outb[row * XS + col] = (_Float16)y;        // in-place, elementwise
      }
    }
  }
  __syncthreads();

  // ---------------- P9: final LN statistics (2 threads/row, v8h reads) -----
  {
    int r = tid >> 1, half = tid & 1;
    const _Float16* rp = &sm.outb[r * XS + half * 128];
    float s = 0.f, s2 = 0.f;
    #pragma unroll
    for (int c = 0; c < 128; c += 8) {
      v8h v = *(const v8h*)(rp + c);
      #pragma unroll
      for (int j = 0; j < 8; ++j) { float f = (float)v[j]; s += f; s2 += f * f; }
    }
    sm.fw[tid * 2]     = s;     // fw region is dead; reuse for partials
    sm.fw[tid * 2 + 1] = s2;
  }
  __syncthreads();
  if (tid < 128) {
    float ss  = sm.fw[tid * 4]     + sm.fw[tid * 4 + 2];
    float ss2 = sm.fw[tid * 4 + 1] + sm.fw[tid * 4 + 3];
    float m  = ss * (1.0f / 256.0f);
    float v2 = ss2 * (1.0f / 256.0f) - m * m;
    sm.muS[tid] = m;
    sm.rsS[tid] = rsqrtf(v2 + EPS_LN);
  }
  __syncthreads();

  // ---------------- P10: normalize + coalesced float4 store ----------------
  {
    float4* ob4 = (float4*)(out + (size_t)rowbase * 256);
    for (int i = 0; i < 32; ++i) {
      int e = i * 256 + tid;                 // float4 index
      int r = e >> 6, c4 = (e & 63) << 2;
      float mu = sm.muS[r], rs = sm.rsS[r];
      const _Float16* p = &sm.outb[r * XS + c4];
      float4 v;
      v.x = ((float)p[0] - mu) * rs * sm.lnfgS[c4]     + sm.lnfbS[c4];
      v.y = ((float)p[1] - mu) * rs * sm.lnfgS[c4 + 1] + sm.lnfbS[c4 + 1];
      v.z = ((float)p[2] - mu) * rs * sm.lnfgS[c4 + 2] + sm.lnfbS[c4 + 2];
      v.w = ((float)p[3] - mu) * rs * sm.lnfgS[c4 + 3] + sm.lnfbS[c4 + 3];
      ob4[e] = v;
    }
  }
}

extern "C" void kernel_launch(void* const* d_in, const int* in_sizes, int n_in,
                              void* d_out, int out_size, void* d_ws, size_t ws_size,
                              hipStream_t stream) {
  (void)n_in; (void)out_size;
  const float* x      = (const float*)d_in[0];
  const float* mp     = (const float*)d_in[1];
  const float* Wq     = (const float*)d_in[2];
  const float* bq     = (const float*)d_in[3];
  const float* pk     = (const float*)d_in[4];
  const float* pv     = (const float*)d_in[5];
  const float* pimp   = (const float*)d_in[6];
  const float* falpha = (const float*)d_in[7];
  const float* ftemp  = (const float*)d_in[8];
  const float* Wo     = (const float*)d_in[9];
  const float* bo     = (const float*)d_in[10];
  const float* lng    = (const float*)d_in[11];
  const float* lnb    = (const float*)d_in[12];
  const float* Wf     = (const float*)d_in[13];
  const float* bf     = (const float*)d_in[14];
  const float* Wg1    = (const float*)d_in[15];
  const float* bg1    = (const float*)d_in[16];
  const float* Wg2    = (const float*)d_in[17];
  const float* bg2    = (const float*)d_in[18];
  const float* lnfg   = (const float*)d_in[19];
  const float* lnfb   = (const float*)d_in[20];
  float* out = (float*)d_out;

  const int B = in_sizes[1];        // minority_prob has B elements
  const int nblocks = B / 128;      // 2048 for B=262144

  if (ws_size >= (size_t)WS_HALFS * sizeof(_Float16)) {
    _Float16* ws = (_Float16*)d_ws;
    prep_weights<<<(WS_HALFS + 255) / 256, 256, 0, stream>>>(Wq, Wf, Wg1, Wg2, ws);
    faiia_fused<1><<<nblocks, 256, 0, stream>>>(
        x, mp, Wq, bq, pk, pv, pimp, falpha, ftemp, Wo, bo, lng, lnb,
        Wf, bf, Wg1, bg1, Wg2, bg2, lnfg, lnfb, ws, out);
  } else {
    faiia_fused<0><<<nblocks, 256, 0, stream>>>(
        x, mp, Wq, bq, pk, pv, pimp, falpha, ftemp, Wo, bo, lng, lnb,
        Wf, bf, Wg1, bg1, Wg2, bg2, lnfg, lnfb, (const _Float16*)d_ws, out);
  }
}
